// GNN_Virtualnode_91070486545221
// MI455X (gfx1250) — matrix operand order repaired and checked by reading, not compiled, not run on hardware
//
#include <hip/hip_runtime.h>

#define EPS 1e-5f

typedef __attribute__((ext_vector_type(16))) __bf16 v16bf;
typedef __attribute__((ext_vector_type(8)))  __bf16 v8bf;
typedef __attribute__((ext_vector_type(8)))  float  v8f;

// ---------------------------------------------------------------------------
// Weight pre-swizzle: row-major f32 [L,K,J] -> bf16 fragments laid out so that
// in the GEMM each lane loads its v16bf B-fragment contiguously (32 bytes).
// Fragment f = kb*NT + nt; lane = {khalf,n}; element i -> W[kb*32+khalf*8+(i&7)+(i>>3)*16][nt*16+n]
// ---------------------------------------------------------------------------
__global__ void prep_w_kernel(const float* __restrict__ W, __bf16* __restrict__ out,
                              int K, int J, int total) {
    int tid = blockIdx.x * 256 + threadIdx.x;
    if (tid >= total) return;
    int per = K * J;
    int l = tid / per;
    int r = tid - l * per;
    int NT = J >> 4;
    int i    = r & 15;
    int lane = (r >> 4) & 31;
    int f    = r >> 9;
    int nt = f % NT;
    int kb = f / NT;
    int n  = lane & 15;
    int kh = lane >> 4;
    int k = kb * 32 + kh * 8 + (i & 7) + ((i >> 3) << 4);
    int j = nt * 16 + n;
    out[tid] = (__bf16)W[(size_t)l * per + (size_t)k * J + j];
}

// h[n] = x_emb1[x[n,0]] + x_emb2[x[n,1]]
__global__ void init_h_kernel(const int* __restrict__ x, const float* __restrict__ e1,
                              const float* __restrict__ e2, float* __restrict__ h, int nrows) {
    int idx = blockIdx.x * 256 + threadIdx.x;
    if (idx >= nrows * 32) return;
    int n = idx >> 5;
    int c = (idx & 31) * 4;
    int i0 = x[2 * n], i1 = x[2 * n + 1];
    float4 a = *(const float4*)(e1 + (size_t)i0 * 128 + c);
    float4 b = *(const float4*)(e2 + (size_t)i1 * 128 + c);
    float4 r; r.x = a.x + b.x; r.y = a.y + b.y; r.z = a.z + b.z; r.w = a.w + b.w;
    *(float4*)(h + (size_t)n * 128 + c) = r;
}

__global__ void vn_init_kernel(const float* __restrict__ vn_emb, float* __restrict__ vn) {
    int idx = blockIdx.x * 256 + threadIdx.x;
    if (idx >= 128 * 128) return;
    vn[idx] = vn_emb[idx & 127];
}

// h[n] += vn[batch[n]]
__global__ void add_vn_kernel(float* __restrict__ h, const float* __restrict__ vn,
                              const int* __restrict__ batch, int nrows) {
    int idx = blockIdx.x * 256 + threadIdx.x;
    if (idx >= nrows * 32) return;
    int n = idx >> 5;
    int c = (idx & 31) * 4;
    int g = batch[n];
    float4 hv = *(const float4*)(h + (size_t)n * 128 + c);
    float4 vv = *(const float4*)(vn + (size_t)g * 128 + c);
    hv.x += vv.x; hv.y += vv.y; hv.z += vv.z; hv.w += vv.w;
    *(float4*)(h + (size_t)n * 128 + c) = hv;
}

// comb[a0*3+a1] = ee1[a0] + ee2[a1]  (18 x 128, per layer; self-loop = comb[12])
__global__ void comb_kernel(const float* __restrict__ ee1l, const float* __restrict__ ee2l,
                            float* __restrict__ comb) {
    int idx = blockIdx.x * 256 + threadIdx.x;
    if (idx >= 18 * 128) return;
    int t = idx >> 7, c = idx & 127;
    comb[idx] = ee1l[(t / 3) * 128 + c] + ee2l[(t % 3) * 128 + c];
}

// ---------------------------------------------------------------------------
// CSR build (once per launch; edge_index / edge_attr are layer-invariant)
// ---------------------------------------------------------------------------
__global__ void zero_int_kernel(int* __restrict__ p, int n) {
    int i = blockIdx.x * 256 + threadIdx.x;
    if (i < n) p[i] = 0;
}

__global__ void edge_count_kernel(const int* __restrict__ ei, int* cnt, int nE) {
    int e = blockIdx.x * 256 + threadIdx.x;
    if (e >= nE) return;
    atomicAdd(&cnt[ei[nE + e]], 1);
}

#define SCAN_CHUNK 1024
__global__ __launch_bounds__(256) void scan_phase1(const int* __restrict__ cnt,
                                                   int* __restrict__ blocksum, int n) {
    __shared__ int sdata[256];
    int base = blockIdx.x * SCAN_CHUNK + threadIdx.x * 4;
    int s = 0;
    #pragma unroll
    for (int i = 0; i < 4; ++i) { int idx = base + i; s += (idx < n) ? cnt[idx] : 0; }
    sdata[threadIdx.x] = s;
    __syncthreads();
    for (int off = 128; off > 0; off >>= 1) {
        if ((int)threadIdx.x < off) sdata[threadIdx.x] += sdata[threadIdx.x + off];
        __syncthreads();
    }
    if (threadIdx.x == 0) blocksum[blockIdx.x] = sdata[0];
}

__global__ void scan_phase2(int* blocksum, int nb) {
    if (threadIdx.x == 0 && blockIdx.x == 0) {
        int acc = 0;
        for (int i = 0; i < nb; ++i) { int v = blocksum[i]; blocksum[i] = acc; acc += v; }
    }
}

__global__ __launch_bounds__(256) void scan_phase3(const int* __restrict__ cnt,
                                                   const int* __restrict__ blocksum,
                                                   int* __restrict__ rowptr,
                                                   int* __restrict__ cursor, int n, int total) {
    __shared__ int sdata[256];
    int base = blockIdx.x * SCAN_CHUNK + threadIdx.x * 4;
    int v[4]; int s = 0;
    #pragma unroll
    for (int i = 0; i < 4; ++i) { int idx = base + i; v[i] = (idx < n) ? cnt[idx] : 0; s += v[i]; }
    sdata[threadIdx.x] = s;
    __syncthreads();
    for (int off = 1; off < 256; off <<= 1) {   // inclusive Hillis-Steele
        int add = ((int)threadIdx.x >= off) ? sdata[threadIdx.x - off] : 0;
        __syncthreads();
        sdata[threadIdx.x] += add;
        __syncthreads();
    }
    int excl = blocksum[blockIdx.x] + sdata[threadIdx.x] - s;
    #pragma unroll
    for (int i = 0; i < 4; ++i) {
        int idx = base + i;
        if (idx < n) { rowptr[idx] = excl; cursor[idx] = excl; }
        excl += v[i];
    }
    if (blockIdx.x == 0 && threadIdx.x == 0) rowptr[n] = total;
}

// CSR payload stored directly: src node + packed edge-emb index (a0*3+a1)
__global__ void edge_fill_kernel(const int* __restrict__ ei, const int* __restrict__ ea,
                                 int* cursor, int* __restrict__ esrc,
                                 int* __restrict__ epack, int nE) {
    int e = blockIdx.x * 256 + threadIdx.x;
    if (e >= nE) return;
    int pos = atomicAdd(&cursor[ei[nE + e]], 1);
    esrc[pos]  = ei[e];
    epack[pos] = ea[2 * e] * 3 + ea[2 * e + 1];
}

// batch is sorted: gstart[g] = first row of graph g, gstart[G] = N
__global__ void graph_bounds_kernel(const int* __restrict__ batch, int* __restrict__ gstart,
                                    int nN, int G) {
    int n = blockIdx.x * 256 + threadIdx.x;
    if (n >= nN) return;
    int g  = batch[n];
    int gp = (n == 0) ? -1 : batch[n - 1];
    for (int gg = gp + 1; gg <= g; ++gg) gstart[gg] = n;
    if (n == nN - 1) { for (int gg = g + 1; gg <= G; ++gg) gstart[gg] = nN; }
}

// ---------------------------------------------------------------------------
// Gather aggregation (no f32 atomics): one wave per node, lane owns 4 columns.
// agg[n] = h[n] + comb[12] + sum_{e: dst==n} (h[src_e] + comb[pack_e])
// ---------------------------------------------------------------------------
__global__ __launch_bounds__(256) void gather_kernel(
    const float* __restrict__ h, const int* __restrict__ esrc, const int* __restrict__ epack,
    const int* __restrict__ rowptr, const float* __restrict__ comb,
    float* __restrict__ agg, int nN) {
    int wid  = (blockIdx.x * 256 + threadIdx.x) >> 5;
    int lane = threadIdx.x & 31;
    if (wid >= nN) return;
    int c = lane * 4;
    int beg = __builtin_amdgcn_readfirstlane(rowptr[wid]);
    int end = __builtin_amdgcn_readfirstlane(rowptr[wid + 1]);
    float4 acc = *(const float4*)(h + (size_t)wid * 128 + c);
    float4 sl  = *(const float4*)(comb + 12 * 128 + c);      // self-loop ee1[4]+ee2[0]
    acc.x += sl.x; acc.y += sl.y; acc.z += sl.z; acc.w += sl.w;
    for (int p = beg; p < end; ++p) {
        int s  = __builtin_amdgcn_readfirstlane(esrc[p]);
        int pk = __builtin_amdgcn_readfirstlane(epack[p]);
        if (p + 1 < end) {   // prefetch next source row (global_prefetch path)
            int sn = __builtin_amdgcn_readfirstlane(esrc[p + 1]);
            __builtin_prefetch(h + (size_t)sn * 128 + c, 0, 1);
        }
        float4 hv = *(const float4*)(h + (size_t)s * 128 + c);
        float4 ec = *(const float4*)(comb + (size_t)pk * 128 + c);
        acc.x += hv.x + ec.x; acc.y += hv.y + ec.y;
        acc.z += hv.z + ec.z; acc.w += hv.w + ec.w;
    }
    *(float4*)(agg + (size_t)wid * 128 + c) = acc;
}

// ---------------------------------------------------------------------------
// Fused node MLP:  h2 = relu(agg @ W1 + b1) @ W2 + b2
// One wave per 16-row strip; bf16 WMMA with f32 accumulation.
// h2 may alias agg: each wave fully loads its agg rows into registers first.
// ---------------------------------------------------------------------------
__global__ __launch_bounds__(128) void gin_mlp_kernel(
    const float* agg, float* h2,
    const __bf16* __restrict__ w1s, const __bf16* __restrict__ w2s,
    const float* __restrict__ b1, const float* __restrict__ b2, int nrows) {
    __shared__ __bf16 tbuf[4][16][264];   // 264 = 256 + pad (LDS bank spread)
    const int wave = threadIdx.x >> 5;
    const int lane = threadIdx.x & 31;
    const int strip = blockIdx.x * 4 + wave;
    const bool active = (strip * 16 < nrows);
    const int m  = lane & 15;
    const int kh = lane >> 4;

    v16bf afrag[4];
    if (active) {
        const int row = strip * 16 + m;
        #pragma unroll
        for (int kb = 0; kb < 4; ++kb) {
            const float* p = agg + (size_t)row * 128 + kb * 32 + kh * 8;
            float4 x0 = *(const float4*)(p);
            float4 x1 = *(const float4*)(p + 4);
            float4 x2 = *(const float4*)(p + 16);
            float4 x3 = *(const float4*)(p + 20);
            v16bf a;
            a[0]  = (__bf16)x0.x; a[1]  = (__bf16)x0.y; a[2]  = (__bf16)x0.z; a[3]  = (__bf16)x0.w;
            a[4]  = (__bf16)x1.x; a[5]  = (__bf16)x1.y; a[6]  = (__bf16)x1.z; a[7]  = (__bf16)x1.w;
            a[8]  = (__bf16)x2.x; a[9]  = (__bf16)x2.y; a[10] = (__bf16)x2.z; a[11] = (__bf16)x2.w;
            a[12] = (__bf16)x3.x; a[13] = (__bf16)x3.y; a[14] = (__bf16)x3.z; a[15] = (__bf16)x3.w;
            afrag[kb] = a;
        }
        // GEMM1: [16x128] x [128x256], relu -> LDS (bf16)
        for (int nt = 0; nt < 16; ++nt) {
            v8f c = {};
            #pragma unroll
            for (int kb = 0; kb < 4; ++kb) {
                v16bf b = *(const v16bf*)(w1s + (((size_t)(kb * 16 + nt) * 32 + lane) << 4));
                c = __builtin_amdgcn_wmma_f32_16x16x32_bf16(false, afrag[kb], false, b,
                                                            (short)0, c, false, false);
            }
            float bias = b1[nt * 16 + m];
            #pragma unroll
            for (int r = 0; r < 8; ++r) {
                float v = c[r] + bias;
                tbuf[wave][r + kh * 8][nt * 16 + m] = (__bf16)(v > 0.f ? v : 0.f);
            }
        }
    }
    __syncthreads();
    if (active) {
        // GEMM2: [16x256] x [256x128] -> h2 (f32, biased; BN applied later)
        for (int nt = 0; nt < 8; ++nt) {
            v8f c = {};
            #pragma unroll
            for (int kb = 0; kb < 8; ++kb) {
                const __bf16* tp = &tbuf[wave][m][kb * 32 + kh * 8];
                v8bf lo = *(const v8bf*)(tp);
                v8bf hi = *(const v8bf*)(tp + 16);
                v16bf a;
                #pragma unroll
                for (int i = 0; i < 8; ++i) { a[i] = lo[i]; a[8 + i] = hi[i]; }
                v16bf b = *(const v16bf*)(w2s + (((size_t)(kb * 8 + nt) * 32 + lane) << 4));
                c = __builtin_amdgcn_wmma_f32_16x16x32_bf16(false, a, false, b,
                                                            (short)0, c, false, false);
            }
            float bias = b2[nt * 16 + m];
            #pragma unroll
            for (int r = 0; r < 8; ++r) {
                h2[(size_t)(strip * 16 + r + kh * 8) * 128 + nt * 16 + m] = c[r] + bias;
            }
        }
    }
}

__global__ void zero_kernel(float* __restrict__ p, int n) {
    int i = blockIdx.x * 256 + threadIdx.x;
    if (i < n) p[i] = 0.f;
}

// Column sums / sums of squares into stats[0..127]=sum, stats[128..255]=sumsq
__global__ __launch_bounds__(128) void bn_stats_kernel(const float* __restrict__ h2,
                                                       float* stats, int nrows) {
    int c = threadIdx.x;
    float s = 0.f, q = 0.f;
    for (int r = blockIdx.x; r < nrows; r += gridDim.x) {
        float v = h2[(size_t)r * 128 + c];
        s += v; q += v * v;
    }
    atomicAdd(&stats[c], s);
    atomicAdd(&stats[128 + c], q);
}

__global__ void bn_apply_kernel(const float* __restrict__ h2, float* __restrict__ out,
                                const float* __restrict__ stats, const float* __restrict__ g,
                                const float* __restrict__ b, int nrows, int relu) {
    int idx = blockIdx.x * 256 + threadIdx.x;
    if (idx >= nrows * 128) return;
    int c = idx & 127;
    float inv = 1.f / (float)nrows;
    float mu  = stats[c] * inv;
    float var = stats[128 + c] * inv - mu * mu;
    float v = g[c] * (h2[idx] - mu) * rsqrtf(var + EPS) + b[c];
    if (relu) v = v > 0.f ? v : 0.f;
    out[idx] = v;
}

// pooled[g] = vn[g] + sum_{rows of graph g} h[row]   (batch sorted -> contiguous)
__global__ __launch_bounds__(128) void pool_gather_kernel(
    const float* __restrict__ h, const int* __restrict__ gstart,
    const float* __restrict__ vn, float* __restrict__ pooled) {
    int g = blockIdx.x;
    int c = threadIdx.x;
    int beg = gstart[g], end = gstart[g + 1];
    float acc = vn[g * 128 + c];
    for (int r = beg; r < end; ++r) acc += h[(size_t)r * 128 + c];
    pooled[g * 128 + c] = acc;
}

// tiny dense: out[g][j] = sum_k in[g][k]*W[k][j] + bias[j]  (G=128 rows)
__global__ void vn_lin_kernel(const float* __restrict__ in, const float* __restrict__ W,
                              const float* __restrict__ bias, float* __restrict__ out,
                              int K, int J) {
    int idx = blockIdx.x * 256 + threadIdx.x;
    if (idx >= 128 * J) return;
    int g = idx / J, j = idx - g * J;
    float acc = bias[j];
    for (int k = 0; k < K; ++k) acc += in[(size_t)g * K + k] * W[(size_t)k * J + j];
    out[idx] = acc;
}

// in-place BN (over 128 rows) + relu, one block of J threads
__global__ void vn_bn_relu_kernel(float* t, const float* __restrict__ g,
                                  const float* __restrict__ b, int J) {
    int j = threadIdx.x;
    if (j >= J) return;
    float s = 0.f, q = 0.f;
    for (int r = 0; r < 128; ++r) { float v = t[(size_t)r * J + j]; s += v; q += v * v; }
    float mu  = s * (1.f / 128.f);
    float var = q * (1.f / 128.f) - mu * mu;
    float sc  = g[j] * rsqrtf(var + EPS);
    float bb  = b[j];
    for (int r = 0; r < 128; ++r) {
        float v = sc * (t[(size_t)r * J + j] - mu) + bb;
        t[(size_t)r * J + j] = v > 0.f ? v : 0.f;
    }
}

extern "C" void kernel_launch(void* const* d_in, const int* in_sizes, int n_in,
                              void* d_out, int out_size, void* d_ws, size_t ws_size,
                              hipStream_t stream) {
    const int N = 100000, E = 1600000, D = 128, L = 5, G = 128;
    const int*   x      = (const int*)d_in[0];
    const int*   ei     = (const int*)d_in[1];
    const int*   ea     = (const int*)d_in[2];
    const int*   batch  = (const int*)d_in[3];
    const float* x_emb1 = (const float*)d_in[4];
    const float* x_emb2 = (const float*)d_in[5];
    const float* vn_emb = (const float*)d_in[6];
    const float* gin_w1 = (const float*)d_in[7];
    const float* gin_b1 = (const float*)d_in[8];
    const float* gin_w2 = (const float*)d_in[9];
    const float* gin_b2 = (const float*)d_in[10];
    const float* ee1    = (const float*)d_in[11];
    const float* ee2    = (const float*)d_in[12];
    const float* bn_g   = (const float*)d_in[13];
    const float* bn_b   = (const float*)d_in[14];
    const float* vn_w1  = (const float*)d_in[15];
    const float* vn_b1  = (const float*)d_in[16];
    const float* vn_bn1_g = (const float*)d_in[17];
    const float* vn_bn1_b = (const float*)d_in[18];
    const float* vn_w2  = (const float*)d_in[19];
    const float* vn_b2  = (const float*)d_in[20];
    const float* vn_bn2_g = (const float*)d_in[21];
    const float* vn_bn2_b = (const float*)d_in[22];

    char* ws = (char*)d_ws;
    const size_t NB = (size_t)N * D * sizeof(float);
    size_t off = 0;
    float*  hA     = (float*)(ws + off); off += NB;
    float*  hB     = (float*)(ws + off); off += NB;
    float*  agg    = (float*)(ws + off); off += NB;        // doubles as h2
    float*  pooled = (float*)(ws + off); off += 65536;
    float*  vn     = (float*)(ws + off); off += 65536;
    float*  tvn    = (float*)(ws + off); off += 131072;
    float*  stats  = (float*)(ws + off); off += 4096;
    float*  comb   = (float*)(ws + off); off += 12288;     // 18 x 128 f32
    __bf16* w1s    = (__bf16*)(ws + off); off += (size_t)L * D * 2 * D * sizeof(__bf16);
    __bf16* w2s    = (__bf16*)(ws + off); off += (size_t)L * D * 2 * D * sizeof(__bf16);
    int*    ecnt     = (int*)(ws + off); off += 400384;
    int*    erowptr  = (int*)(ws + off); off += 400384;    // N+1 ints
    int*    ecursor  = (int*)(ws + off); off += 400384;
    int*    esrc     = (int*)(ws + off); off += (size_t)E * sizeof(int);
    int*    epack    = (int*)(ws + off); off += (size_t)E * sizeof(int);
    int*    blocksum = (int*)(ws + off); off += 1024;
    int*    gstart   = (int*)(ws + off); off += 1024;      // G+1 ints

    // --- one-time per call: pre-swizzle GIN weights into WMMA B-fragment order
    const int totW = L * D * (2 * D);
    prep_w_kernel<<<(totW + 255) / 256, 256, 0, stream>>>(gin_w1, w1s, D, 2 * D, totW);
    prep_w_kernel<<<(totW + 255) / 256, 256, 0, stream>>>(gin_w2, w2s, 2 * D, D, totW);

    // --- one-time per call: CSR by destination (layer-invariant) + graph bounds
    const int NBLK = (N + SCAN_CHUNK - 1) / SCAN_CHUNK;
    zero_int_kernel<<<(N + 255) / 256, 256, 0, stream>>>(ecnt, N);
    edge_count_kernel<<<(E + 255) / 256, 256, 0, stream>>>(ei, ecnt, E);
    scan_phase1<<<NBLK, 256, 0, stream>>>(ecnt, blocksum, N);
    scan_phase2<<<1, 1, 0, stream>>>(blocksum, NBLK);
    scan_phase3<<<NBLK, 256, 0, stream>>>(ecnt, blocksum, erowptr, ecursor, N, E);
    edge_fill_kernel<<<(E + 255) / 256, 256, 0, stream>>>(ei, ea, ecursor, esrc, epack, E);
    graph_bounds_kernel<<<(N + 255) / 256, 256, 0, stream>>>(batch, gstart, N, G);

    init_h_kernel<<<(N * 32 + 255) / 256, 256, 0, stream>>>(x, x_emb1, x_emb2, hA, N);
    vn_init_kernel<<<(G * D + 255) / 256, 256, 0, stream>>>(vn_emb, vn);

    float* hCur = hA;
    float* hNext = hB;
    for (int l = 0; l < L; ++l) {
        const float* ee1l = ee1 + (size_t)l * 6 * D;
        const float* ee2l = ee2 + (size_t)l * 3 * D;

        add_vn_kernel<<<(N * 32 + 255) / 256, 256, 0, stream>>>(hCur, vn, batch, N);
        comb_kernel<<<(18 * 128 + 255) / 256, 256, 0, stream>>>(ee1l, ee2l, comb);
        gather_kernel<<<(N * 32 + 255) / 256, 256, 0, stream>>>(
            hCur, esrc, epack, erowptr, comb, agg, N);

        gin_mlp_kernel<<<(N / 16 + 3) / 4, 128, 0, stream>>>(
            agg, agg,
            w1s + (size_t)l * D * 2 * D, w2s + (size_t)l * 2 * D * D,
            gin_b1 + (size_t)l * 2 * D, gin_b2 + (size_t)l * D, N);

        zero_kernel<<<1, 256, 0, stream>>>(stats, 2 * D);
        bn_stats_kernel<<<512, 128, 0, stream>>>(agg, stats, N);

        if (l < L - 1) {  // virtual node pooling uses pre-conv h (still in hCur)
            pool_gather_kernel<<<G, 128, 0, stream>>>(hCur, gstart, vn, pooled);
        }

        float* dst = (l == L - 1) ? (float*)d_out : hNext;
        bn_apply_kernel<<<(N * D + 255) / 256, 256, 0, stream>>>(
            agg, dst, stats, bn_g + (size_t)l * D, bn_b + (size_t)l * D, N, (l < L - 1) ? 1 : 0);

        if (l < L - 1) {
            vn_lin_kernel<<<(G * 2 * D + 255) / 256, 256, 0, stream>>>(
                pooled, vn_w1 + (size_t)l * D * 2 * D, vn_b1 + (size_t)l * 2 * D, tvn, D, 2 * D);
            vn_bn_relu_kernel<<<1, 2 * D, 0, stream>>>(tvn, vn_bn1_g + (size_t)l * 2 * D,
                                                       vn_bn1_b + (size_t)l * 2 * D, 2 * D);
            vn_lin_kernel<<<(G * D + 255) / 256, 256, 0, stream>>>(
                tvn, vn_w2 + (size_t)l * 2 * D * D, vn_b2 + (size_t)l * D, vn, 2 * D, D);
            vn_bn_relu_kernel<<<1, D, 0, stream>>>(vn, vn_bn2_g + (size_t)l * D,
                                                   vn_bn2_b + (size_t)l * D, D);
            float* tmp = hCur; hCur = hNext; hNext = tmp;
        }
    }
    (void)in_sizes; (void)n_in; (void)out_size; (void)ws_size;
}
